// PassageDecoder_52570399703645
// MI455X (gfx1250) — matrix operand order repaired and checked by reading, not compiled, not run on hardware
//
#include <hip/hip_runtime.h>
#include <math.h>

// Problem constants (match reference)
#define Bq   32
#define Tq   32
#define Sq   256
#define Vq   32000
#define Eq   256
#define Hq   512
#define OOVq 50
#define VEXTq (Vq + OOVq)
#define NEGq 1.0e12f

typedef __attribute__((ext_vector_type(16))) __bf16 v16bf;
typedef __attribute__((ext_vector_type(8)))  float  v8f;

// ---- CDNA5 async global->LDS path (guarded: falls back to sync staging) ----
#if defined(__has_builtin)
#  if __has_builtin(__builtin_amdgcn_global_load_async_to_lds_b128) && \
      __has_builtin(__builtin_amdgcn_global_load_async_to_lds_b32)
#    define HAS_ASYNC_LDS 1
#  endif
#endif
#ifndef HAS_ASYNC_LDS
#  define HAS_ASYNC_LDS 0
#endif

#if HAS_ASYNC_LDS
// Signatures (from hipcc diagnostics): arg0 = global src, arg1 = LDS dst,
// arg2/arg3 = imm offset / cpol.
//   b32 : (__device__ int*,            __shared__ ...)
//   b128: (__device__ int __vector(4)*,__shared__ ...)
typedef int v4i_vs __attribute__((vector_size(16)));
typedef __attribute__((address_space(1))) int*    g_i32p;
typedef __attribute__((address_space(1))) v4i_vs* g_v4ip;
typedef __attribute__((address_space(3))) int*    l_i32p;
typedef __attribute__((address_space(3))) v4i_vs* l_v4ip;
#endif

__device__ __forceinline__ void wait_async_lds() {
#if HAS_ASYNC_LDS
#  if __has_builtin(__builtin_amdgcn_s_wait_asynccnt)
  __builtin_amdgcn_s_wait_asynccnt(0);
#  else
  asm volatile("s_wait_asynccnt 0x0" ::: "memory");
#  endif
#endif
}

// ---------- helpers ----------
__device__ __forceinline__ unsigned short f32_to_bf16u(float f) {
  unsigned u = __float_as_uint(f);
  unsigned r = u + 0x7FFFu + ((u >> 16) & 1u);   // round-to-nearest-even
  return (unsigned short)(r >> 16);
}
__device__ __forceinline__ __bf16 bits_to_bf(unsigned short u) {
  union { unsigned short u; __bf16 b; } v; v.u = u; return v.b;
}
__device__ __forceinline__ float sigmoidf_(float x) { return 1.0f / (1.0f + expf(-x)); }

// order-preserving float<->uint map for atomicMax on floats
__device__ __forceinline__ unsigned f2ord(float f) {
  unsigned u = __float_as_uint(f);
  return (u & 0x80000000u) ? ~u : (u | 0x80000000u);
}
__device__ __forceinline__ float ord2f(unsigned k) {
  unsigned u = (k & 0x80000000u) ? (k & 0x7FFFFFFFu) : ~k;
  return __uint_as_float(u);
}

// ---------- LDS double-buffered bf16 WMMA GEMM ----------
// C[m,n] = act( sum_k A[m,k]*B[k,n] (+ sum_k A2[m,k]*B2[k,n]) + bias[n] )
// A,A2: [M,K] bf16 row-major. B,B2: [K,N] bf16 row-major. C row stride = ldC.
// Block = 256 threads (8 waves) computing a 16x128 C strip; one M-tile shared
// by all waves, each wave owns one 16x16 N-tile. Per 32-K slab, B is staged
// row-major with 16B transfers and A in WMMA-fragment order. With async
// global->LDS ops the next slab streams into the alternate buffer while the
// current slab is consumed (one s_wait_asynccnt + one barrier per slab).
// Requires: M % 16 == 0, N % 128 == 0, K % 32 == 0 (true for all uses here).
__global__ void __launch_bounds__(256) k_wmma_gemm(
    const unsigned short* __restrict__ A,
    const unsigned short* __restrict__ Bm, int K,
    const unsigned short* __restrict__ A2,
    const unsigned short* __restrict__ B2, int K2,
    const float* __restrict__ bias,
    float* __restrict__ Cf, unsigned short* __restrict__ Cb,
    int M, int N, long long ldC, int act) {
  __shared__ __align__(32) unsigned short ldsA[2][16 * 32];    // fragment-order A slabs
  __shared__ __align__(16) unsigned short ldsB[2][32 * 128];   // row-major B slabs

  const int tid  = threadIdx.x;
  const int lane = tid & 31;
  const int wave = tid >> 5;                 // 0..7: N-tile within the strip
  const int stripsN = N >> 7;
  const int row0    = (blockIdx.x / stripsN) << 4;
  const int colBase = (blockIdx.x % stripsN) << 7;
  const int col0    = colBase + (wave << 4);

  // CDNA5 wave32 WMMA fragment geometry (CDNA5 ISA)
  const int nB  = lane & 15;
  const int kbB = (lane >> 4) << 4;          // lanes 16-31 carry K+16 rows of B

  // A staging decomposition: thread -> (fragment lane l, uint pair p)
  const int sl = tid >> 3;                   // 0..31
  const int sp = tid & 7;                    // 0..7
  const int skpair = (2 * sp) + ((sl >= 16) ? 16 : 0);
  const int sm = sl & 15;

  const int s1 = K >> 5;
  const int nslab = s1 + (A2 ? (K2 >> 5) : 0);

  // stage K-slab s into LDS buffer bufIdx
  auto stage = [&](int s, int bufIdx) {
    const unsigned short* Ap; const unsigned short* Bp; int Kp, kk;
    if (s < s1) { Ap = A;  Bp = Bm; Kp = K;  kk = s << 5; }
    else        { Ap = A2; Bp = B2; Kp = K2; kk = (s - s1) << 5; }
    unsigned short* lA = ldsA[bufIdx];
    unsigned short* lB = ldsB[bufIdx];
    const unsigned short* srcA = Ap + (size_t)(row0 + sm) * (size_t)Kp + kk + skpair;
#if HAS_ASYNC_LDS
    __builtin_amdgcn_global_load_async_to_lds_b32(
        (g_i32p)srcA, (l_i32p)(((unsigned*)lA) + tid), 0, 0);
#else
    ((unsigned*)lA)[tid] = *(const unsigned*)srcA;
#endif
#pragma unroll
    for (int c2 = tid; c2 < 512; c2 += 256) {
      int krow = c2 >> 4;
      int cg   = (c2 & 15) << 3;             // ushort group of 8 (16B)
      const unsigned short* srcB = Bp + (size_t)(kk + krow) * (size_t)N + colBase + cg;
#if HAS_ASYNC_LDS
      __builtin_amdgcn_global_load_async_to_lds_b128(
          (g_v4ip)srcB, (l_v4ip)(((uint4*)lB) + c2), 0, 0);
#else
      ((uint4*)lB)[c2] = *(const uint4*)srcB;
#endif
    }
#if !HAS_ASYNC_LDS
    if (kk + 32 < Kp)                        // keep next slab warm in L2
      __builtin_prefetch(Bp + (size_t)(kk + 32) * (size_t)N + colBase + (tid & 15) * 8, 0, 1);
#endif
  };

  v8f acc = {0.f, 0.f, 0.f, 0.f, 0.f, 0.f, 0.f, 0.f};

  stage(0, 0);
  for (int s = 0; s < nslab; ++s) {
    wait_async_lds();                        // own slab-s transfers complete
    __syncthreads();                         // all waves' slab-s data visible;
                                             // slab-(s-1) consumers done
    if (s + 1 < nslab) stage(s + 1, (s + 1) & 1);

    const unsigned short* lA = ldsA[s & 1];
    const unsigned short* lB = ldsB[s & 1];
    // A fragment: contiguous 32B per lane -> ds_load_b128 x2
    v16bf afrag = *(const v16bf*)(lA + lane * 16);
    // B fragment: 16 u16 LDS reads (cheap 32-bit address math)
    v16bf bfrag;
#pragma unroll
    for (int j = 0; j < 16; ++j)
      bfrag[j] = bits_to_bf(lB[(kbB + j) * 128 + (wave << 4) + nB]);

    acc = __builtin_amdgcn_wmma_f32_16x16x32_bf16(
        false, afrag, false, bfrag, (short)0, acc, false, false);
  }

  const float bcol = bias ? bias[col0 + nB] : 0.f;
  const int mo = (lane >> 4) << 3;           // VGPR r: M=r (lanes 0-15) / M=8+r (lanes 16-31)
#pragma unroll
  for (int r = 0; r < 8; ++r) {
    float val = acc[r] + bcol;
    if (act == 1) val = tanhf(val);
    size_t o = (size_t)(row0 + r + mo) * (size_t)ldC + (size_t)(col0 + nB);
    if (Cf) Cf[o] = val;
    if (Cb) Cb[o] = f32_to_bf16u(val);
  }
}

// ---------- small kernels ----------
__global__ void k_cvt_bf16(const float* __restrict__ s, unsigned short* __restrict__ d, int n) {
  int i = blockIdx.x * blockDim.x + threadIdx.x;
  if (i < n) d[i] = f32_to_bf16u(s[i]);
}

__global__ void k_init_state(const float* __restrict__ h0, const float* __restrict__ c0,
                             float* __restrict__ h, float* __restrict__ c,
                             float* __restrict__ ctx, unsigned short* __restrict__ hbf) {
  int i = blockIdx.x * blockDim.x + threadIdx.x;
  if (i < Bq * Hq) { h[i] = h0[i]; c[i] = c0[i]; ctx[i] = 0.f; hbf[i] = f32_to_bf16u(h0[i]); }
}

// first-occurrence ownership per (b, s) over ext_src (fixed across timesteps)
__global__ void k_owner(const int* __restrict__ ext_src, int* __restrict__ owner) {
  int i = blockIdx.x * blockDim.x + threadIdx.x;
  if (i >= Bq * Sq) return;
  int b = i / Sq, s = i % Sq, idx = ext_src[i], own = 1;
  for (int j = 0; j < s; ++j)
    if (ext_src[b * Sq + j] == idx) { own = 0; break; }
  owner[i] = own;
}

__global__ void k_build_xin(const int* __restrict__ trg, int t, const float* __restrict__ emb,
                            const float* __restrict__ ctx, unsigned short* __restrict__ xin) {
  const int W = Eq + Hq;
  int i = blockIdx.x * blockDim.x + threadIdx.x;
  if (i >= Bq * W) return;
  int b = i / W, j = i % W;
  float v = (j < Eq) ? emb[(size_t)trg[b * Tq + t] * Eq + j] : ctx[b * Hq + (j - Eq)];
  xin[i] = f32_to_bf16u(v);
}

__global__ void k_lstm(const float* __restrict__ gates, float* __restrict__ h,
                       float* __restrict__ c, unsigned short* __restrict__ hbf) {
  int i = blockIdx.x * blockDim.x + threadIdx.x;
  if (i >= Bq * Hq) return;
  int b = i / Hq, j = i % Hq;
  const float* g = gates + (size_t)b * 4 * Hq;
  float ig = sigmoidf_(g[j]);
  float fg = sigmoidf_(g[Hq + j]);
  float gg = tanhf(g[2 * Hq + j]);
  float og = sigmoidf_(g[3 * Hq + j]);
  float cn = fg * c[i] + ig * gg;
  float hn = og * tanhf(cn);
  c[i] = cn; h[i] = hn; hbf[i] = f32_to_bf16u(hn);
}

// energy[b,s] = dot(h[b], memories[b,s]) - NEG*(1-mask); one wave32 per (b,s), fp32
__global__ void k_energy(const float* __restrict__ h, const float* __restrict__ mem,
                         const float* __restrict__ mask, float* __restrict__ energy) {
  int gid = blockIdx.x * blockDim.x + threadIdx.x;
  int w = gid >> 5, lane = gid & 31;
  if (w >= Bq * Sq) return;
  int b = w / Sq;
  const float* mrow = mem + (size_t)w * Hq;
  const float* hb = h + (size_t)b * Hq;
  float acc = 0.f;
  for (int j = lane; j < Hq; j += 32) acc += hb[j] * mrow[j];
  for (int off = 16; off; off >>= 1) acc += __shfl_xor(acc, off, 32);
  if (lane == 0) energy[w] = acc - NEGq * (1.0f - mask[w]);
}

// softmax over S=256, ctx = attn @ memories, and pack cat_in = [h, ctx] as bf16.
// blockDim must be 256; one block per batch row.
__global__ void k_softmax_ctx(const float* __restrict__ energy, const float* __restrict__ mem,
                              const float* __restrict__ h, float* __restrict__ ctx,
                              unsigned short* __restrict__ catin) {
  __shared__ float red[256];
  __shared__ float attn[Sq];
  int b = blockIdx.x, tid = threadIdx.x;
  float e = energy[b * Sq + tid];
  red[tid] = e; __syncthreads();
  for (int st = 128; st; st >>= 1) { if (tid < st) red[tid] = fmaxf(red[tid], red[tid + st]); __syncthreads(); }
  float mx = red[0]; __syncthreads();
  float ex = expf(e - mx);
  red[tid] = ex; __syncthreads();
  for (int st = 128; st; st >>= 1) { if (tid < st) red[tid] += red[tid + st]; __syncthreads(); }
  attn[tid] = ex / red[0];
  __syncthreads();
  const float* mb = mem + (size_t)b * Sq * Hq;
  float a0 = 0.f, a1 = 0.f;
  for (int s = 0; s < Sq; ++s) {
    float a = attn[s];
    a0 += a * mb[(size_t)s * Hq + tid];
    a1 += a * mb[(size_t)s * Hq + tid + 256];
  }
  ctx[b * Hq + tid] = a0;
  ctx[b * Hq + tid + 256] = a1;
  unsigned short* cb = catin + (size_t)b * 2 * Hq;
  cb[tid]            = f32_to_bf16u(h[b * Hq + tid]);
  cb[tid + 256]      = f32_to_bf16u(h[b * Hq + tid + 256]);
  cb[Hq + tid]       = f32_to_bf16u(a0);
  cb[Hq + tid + 256] = f32_to_bf16u(a1);
}

// reset touched copy slots to -NEG and zero the OOV tail of out[:, t, V:]
__global__ void k_copy_reset(const int* __restrict__ ext_src, unsigned* __restrict__ cbuf,
                             float* __restrict__ out, int t) {
  int i = blockIdx.x * blockDim.x + threadIdx.x;
  if (i < Bq * Sq) {
    int b = i / Sq;
    cbuf[(size_t)b * VEXTq + ext_src[i]] = f2ord(-NEGq);
  } else {
    int j = i - Bq * Sq;
    if (j < Bq * OOVq) {
      int b = j / OOVq, k = j % OOVq;
      out[((size_t)b * Tq + t) * VEXTq + Vq + k] = 0.f;
    }
  }
}

__global__ void k_copy_scatter(const int* __restrict__ ext_src, const float* __restrict__ energy,
                               unsigned* __restrict__ cbuf) {
  int i = blockIdx.x * blockDim.x + threadIdx.x;
  if (i >= Bq * Sq) return;
  int b = i / Sq;
  atomicMax(&cbuf[(size_t)b * VEXTq + ext_src[i]], f2ord(energy[i]));
}

__global__ void k_copy_apply(const int* __restrict__ ext_src, const int* __restrict__ owner,
                             const unsigned* __restrict__ cbuf, float* __restrict__ out, int t) {
  int i = blockIdx.x * blockDim.x + threadIdx.x;
  if (i >= Bq * Sq) return;
  if (!owner[i]) return;
  int b = i / Sq, idx = ext_src[i];
  float val = ord2f(cbuf[(size_t)b * VEXTq + idx]);
  if (val > -0.5f * NEGq)
    out[((size_t)b * Tq + t) * VEXTq + idx] += val;
}

// ---------- host ----------
static inline int cdiv(int a, int b) { return (a + b - 1) / b; }

extern "C" void kernel_launch(void* const* d_in, const int* in_sizes, int n_in,
                              void* d_out, int out_size, void* d_ws, size_t ws_size,
                              hipStream_t stream) {
  (void)in_sizes; (void)n_in; (void)out_size; (void)ws_size;
  const int*   trg   = (const int*)d_in[0];
  const int*   ext   = (const int*)d_in[1];
  const float* enco  = (const float*)d_in[2];
  const float* mask  = (const float*)d_in[3];
  const float* h0    = (const float*)d_in[4];
  const float* c0    = (const float*)d_in[5];
  const float* emb   = (const float*)d_in[6];
  const float* encW  = (const float*)d_in[7];
  const float* encb  = (const float*)d_in[8];
  const float* redW  = (const float*)d_in[9];
  const float* redb  = (const float*)d_in[10];
  const float* Wx    = (const float*)d_in[11];
  const float* Wh    = (const float*)d_in[12];
  const float* blstm = (const float*)d_in[13];
  const float* catW  = (const float*)d_in[14];
  const float* catb  = (const float*)d_in[15];
  const float* logW  = (const float*)d_in[16];
  const float* logb  = (const float*)d_in[17];
  float* out = (float*)d_out;

  // workspace carve-out (~68 MB total)
  char* ws = (char*)d_ws;
  size_t off = 0;
  auto alloc = [&](size_t bytes) -> char* {
    char* p = ws + off; off = (off + bytes + 255) & ~(size_t)255; return p;
  };
  const int BSH = Bq * Sq * Hq;
  float*          memories = (float*)         alloc((size_t)BSH * 4);
  unsigned short* encoB    = (unsigned short*)alloc((size_t)BSH * 2);
  unsigned short* encWb    = (unsigned short*)alloc((size_t)Hq * Hq * 2);
  unsigned short* redWb    = (unsigned short*)alloc((size_t)(Eq + Hq) * Eq * 2);
  unsigned short* WxB      = (unsigned short*)alloc((size_t)Eq * 4 * Hq * 2);
  unsigned short* WhB      = (unsigned short*)alloc((size_t)Hq * 4 * Hq * 2);
  unsigned short* catWb    = (unsigned short*)alloc((size_t)2 * Hq * Hq * 2);
  unsigned short* logWb    = (unsigned short*)alloc((size_t)Hq * Vq * 2);
  float*          hS       = (float*)         alloc((size_t)Bq * Hq * 4);
  float*          cS       = (float*)         alloc((size_t)Bq * Hq * 4);
  float*          ctxS     = (float*)         alloc((size_t)Bq * Hq * 4);
  unsigned short* hbf      = (unsigned short*)alloc((size_t)Bq * Hq * 2);
  unsigned short* xin      = (unsigned short*)alloc((size_t)Bq * (Eq + Hq) * 2);
  unsigned short* xr       = (unsigned short*)alloc((size_t)Bq * Eq * 2);
  float*          gates    = (float*)         alloc((size_t)Bq * 4 * Hq * 4);
  unsigned short* catin    = (unsigned short*)alloc((size_t)Bq * 2 * Hq * 2);
  unsigned short* feat     = (unsigned short*)alloc((size_t)Bq * Hq * 2);
  float*          energy   = (float*)         alloc((size_t)Bq * Sq * 4);
  unsigned*       cbuf     = (unsigned*)      alloc((size_t)Bq * VEXTq * 4);
  int*            owner    = (int*)           alloc((size_t)Bq * Sq * 4);

  const int TH = 256;
  // one-time bf16 conversions (weights then stay resident in the 192MB L2)
  k_cvt_bf16<<<cdiv(BSH, TH), TH, 0, stream>>>(enco, encoB, BSH);
  k_cvt_bf16<<<cdiv(Hq * Hq, TH), TH, 0, stream>>>(encW, encWb, Hq * Hq);
  k_cvt_bf16<<<cdiv((Eq + Hq) * Eq, TH), TH, 0, stream>>>(redW, redWb, (Eq + Hq) * Eq);
  k_cvt_bf16<<<cdiv(Eq * 4 * Hq, TH), TH, 0, stream>>>(Wx, WxB, Eq * 4 * Hq);
  k_cvt_bf16<<<cdiv(Hq * 4 * Hq, TH), TH, 0, stream>>>(Wh, WhB, Hq * 4 * Hq);
  k_cvt_bf16<<<cdiv(2 * Hq * Hq, TH), TH, 0, stream>>>(catW, catWb, 2 * Hq * Hq);
  k_cvt_bf16<<<cdiv(Hq * Vq, TH), TH, 0, stream>>>(logW, logWb, Hq * Vq);

  // memories = enc_out @ enc_W + enc_b   (WMMA, [8192,512]x[512,512])
  k_wmma_gemm<<<(Bq * Sq / 16) * (Hq / 128), 256, 0, stream>>>(
      encoB, encWb, Hq, nullptr, nullptr, 0, encb,
      memories, nullptr, Bq * Sq, Hq, (long long)Hq, 0);
  k_init_state<<<cdiv(Bq * Hq, TH), TH, 0, stream>>>(h0, c0, hS, cS, ctxS, hbf);
  k_owner<<<cdiv(Bq * Sq, TH), TH, 0, stream>>>(ext, owner);

  for (int t = 0; t < Tq; ++t) {
    // x_in = [emb[y_t], ctx]  (bf16)
    k_build_xin<<<cdiv(Bq * (Eq + Hq), TH), TH, 0, stream>>>(trg, t, emb, ctxS, xin);
    // x = x_in @ red_W + red_b                 [32,768]x[768,256]
    k_wmma_gemm<<<(Bq / 16) * (Eq / 128), 256, 0, stream>>>(
        xin, redWb, Eq + Hq, nullptr, nullptr, 0, redb,
        nullptr, xr, Bq, Eq, (long long)Eq, 0);
    // gates = x @ Wx + h @ Wh + b              [32,256]x[256,2048] + [32,512]x[512,2048]
    k_wmma_gemm<<<(Bq / 16) * (4 * Hq / 128), 256, 0, stream>>>(
        xr, WxB, Eq, hbf, WhB, Hq, blstm,
        gates, nullptr, Bq, 4 * Hq, (long long)(4 * Hq), 0);
    // LSTM pointwise (fp32)
    k_lstm<<<cdiv(Bq * Hq, TH), TH, 0, stream>>>(gates, hS, cS, hbf);
    // attention energies (fp32, one wave per (b,s))
    k_energy<<<cdiv(Bq * Sq * 32, TH), TH, 0, stream>>>(hS, memories, mask, energy);
    // softmax + context + cat_in packing
    k_softmax_ctx<<<Bq, 256, 0, stream>>>(energy, memories, hS, ctxS, catin);
    // feat = tanh(cat_in @ cat_W + cat_b)      [32,1024]x[1024,512]
    k_wmma_gemm<<<(Bq / 16) * (Hq / 128), 256, 0, stream>>>(
        catin, catWb, 2 * Hq, nullptr, nullptr, 0, catb,
        nullptr, feat, Bq, Hq, (long long)Hq, 1);
    // logits -> out[:, t, :V] directly         [32,512]x[512,32000]  (dominant GEMM)
    k_wmma_gemm<<<(Bq / 16) * (Vq / 128), 256, 0, stream>>>(
        feat, logWb, Hq, nullptr, nullptr, 0, logb,
        out + (size_t)t * VEXTq, nullptr, Bq, Vq, (long long)Tq * VEXTq, 0);
    // pointer-copy: segment-max of energies into extended vocab
    k_copy_reset<<<cdiv(Bq * Sq + Bq * OOVq, TH), TH, 0, stream>>>(ext, cbuf, out, t);
    k_copy_scatter<<<cdiv(Bq * Sq, TH), TH, 0, stream>>>(ext, energy, cbuf);
    k_copy_apply<<<cdiv(Bq * Sq, TH), TH, 0, stream>>>(ext, owner, cbuf, out, t);
  }
}
